// CrossShapedAttention_89446988906649
// MI455X (gfx1250) — matrix operand order repaired and checked by reading, not compiled, not run on hardware
//
#include <hip/hip_runtime.h>
#include <hip/hip_bf16.h>
#include <cstdint>

// ---------------------------------------------------------------------------
// CDNA5 (gfx1250) implementation of cross-shaped window attention block.
// Matrix math: v_wmma_f32_16x16x32_bf16 (wave32).
// Attention K/V staging: global_load_async_to_lds_b128 with double buffering.
// ---------------------------------------------------------------------------

typedef __attribute__((ext_vector_type(16))) __bf16          v16bf;
typedef __attribute__((ext_vector_type(8)))  float           v8f;
typedef __attribute__((ext_vector_type(8)))  unsigned short  v8u16;

union BFrag {
    v16bf          b;
    unsigned short u[16];
    v8u16          h[2];
};

__device__ __forceinline__ unsigned short f2bf(float f) {
    union { float f; unsigned u; } c; c.f = f;
    unsigned u = c.u;
    unsigned r = (u + 0x7FFFu + ((u >> 16) & 1u)) >> 16;   // RNE
    return (unsigned short)r;
}
__device__ __forceinline__ float bf2f(unsigned short s) {
    union { unsigned u; float f; } c; c.u = ((unsigned)s) << 16;
    return c.f;
}
__device__ __forceinline__ float hswish(float x) {
    float t = fminf(fmaxf(x + 3.0f, 0.0f), 6.0f);
    return x * t * (1.0f / 6.0f);
}

// 64B per-lane async copy: memory -> LDS, tracked by ASYNCcnt (no VGPR data).
// s_wait_dscnt 0 guards WAR against pending DS reads of the target buffer.
__device__ __forceinline__ void stage_async64(unsigned lds_addr, const void* src) {
    asm volatile(
        "s_wait_dscnt 0x0\n\t"
        "global_load_async_to_lds_b128 %0, %1, off\n\t"
        "global_load_async_to_lds_b128 %0, %1, off offset:16\n\t"
        "global_load_async_to_lds_b128 %0, %1, off offset:32\n\t"
        "global_load_async_to_lds_b128 %0, %1, off offset:48"
        :: "v"(lds_addr), "v"(src) : "memory");
}
__device__ __forceinline__ void wait_async_keep4() {
    asm volatile("s_wait_asynccnt 0x4" ::: "memory");
}
__device__ __forceinline__ void wait_async_all() {
    asm volatile("s_wait_asynccnt 0x0" ::: "memory");
}

// ---------------------------------------------------------------------------
// Pre-swizzle weights (KxN, f32) into WMMA B-fragment order, bf16:
//   out[(k/32)*(N/16) + n/16][ (n%16)*32 + (k%32) ]
// ---------------------------------------------------------------------------
__global__ void swizzle_w_kernel(const float* __restrict__ w,
                                 unsigned short* __restrict__ o,
                                 int K, int N) {
    int tid = blockIdx.x * blockDim.x + threadIdx.x;
    if (tid >= K * N) return;
    int k = tid / N, n = tid % N;
    o[((size_t)(k >> 5) * (N >> 4) + (n >> 4)) * 512 + (n & 15) * 32 + (k & 31)]
        = f2bf(w[(size_t)k * N + n]);
}

// ---------------------------------------------------------------------------
// Tiled WMMA GEMM: C[M,N] = A[M,K] * B[K,N] + bias.
// Block = 256 threads = 8 waves; block tile 64x64; wave tile 16x32; K step 32.
// ATYPE: 0 = A is f32 (converted to bf16 in registers), 1 = A is bf16.
// MODE : 0 = qkv epilogue (cols <256 -> q_bf16, cols >=256 -> k_bf16)
//        1 = final projection epilogue (f32 output)
// ---------------------------------------------------------------------------
template<int ATYPE, int MODE>
__global__ __launch_bounds__(256) void gemm_wmma_kernel(
    const void* __restrict__ Aptr, const unsigned short* __restrict__ Bsw,
    const float* __restrict__ bias, int K, int N,
    unsigned short* __restrict__ q_bf, unsigned short* __restrict__ k_bf,
    float* __restrict__ out_f32)
{
    const int gm0  = blockIdx.y * 64;
    const int gn0  = blockIdx.x * 64;
    const int wid  = threadIdx.x >> 5;
    const int lane = threadIdx.x & 31;
    const int wm   = wid & 3, wn = wid >> 2;
    const int n    = lane & 15, hi = lane >> 4;
    const int rowA = gm0 + wm * 16 + n;        // A-fragment row (m = lane&15)
    const int ncol0 = gn0 + wn * 32;

    v8f acc0 = {}, acc1 = {};

    for (int kt = 0; kt < K; kt += 32) {
        BFrag af;
        if (ATYPE == 0) {
            const float* ap = (const float*)Aptr + (size_t)rowA * K + kt;
            const float4 fa = *(const float4*)(ap + hi * 16);
            const float4 fb = *(const float4*)(ap + hi * 16 + 4);
            const float4 fc = *(const float4*)(ap + hi * 16 + 8);
            const float4 fd = *(const float4*)(ap + hi * 16 + 12);
            af.u[0]  = f2bf(fa.x); af.u[1]  = f2bf(fa.y);
            af.u[2]  = f2bf(fa.z); af.u[3]  = f2bf(fa.w);
            af.u[4]  = f2bf(fb.x); af.u[5]  = f2bf(fb.y);
            af.u[6]  = f2bf(fb.z); af.u[7]  = f2bf(fb.w);
            af.u[8]  = f2bf(fc.x); af.u[9]  = f2bf(fc.y);
            af.u[10] = f2bf(fc.z); af.u[11] = f2bf(fc.w);
            af.u[12] = f2bf(fd.x); af.u[13] = f2bf(fd.y);
            af.u[14] = f2bf(fd.z); af.u[15] = f2bf(fd.w);
        } else {
            const unsigned short* ap =
                (const unsigned short*)Aptr + (size_t)rowA * K + kt;
            af.h[0] = *(const v8u16*)(ap + hi * 16);
            af.h[1] = *(const v8u16*)(ap + hi * 16 + 8);
        }
        const size_t bbase = ((size_t)(kt >> 5) * (N >> 4) + (ncol0 >> 4)) * 512;
        BFrag bf0, bf1;
        bf0.h[0] = *(const v8u16*)(Bsw + bbase + n * 32 + hi * 16);
        bf0.h[1] = *(const v8u16*)(Bsw + bbase + n * 32 + hi * 16 + 8);
        bf1.h[0] = *(const v8u16*)(Bsw + bbase + 512 + n * 32 + hi * 16);
        bf1.h[1] = *(const v8u16*)(Bsw + bbase + 512 + n * 32 + hi * 16 + 8);

        acc0 = __builtin_amdgcn_wmma_f32_16x16x32_bf16(
                   false, af.b, false, bf0.b, (short)0, acc0, false, false);
        acc1 = __builtin_amdgcn_wmma_f32_16x16x32_bf16(
                   false, af.b, false, bf1.b, (short)0, acc1, false, false);
    }

#pragma unroll
    for (int jj = 0; jj < 2; ++jj) {
        // The 256-channel q/k boundary is 16-aligned, so this condition is
        // wave-uniform; readfirstlane forces it into an SGPR so the q/k split
        // compiles to a scalar branch instead of per-lane exec masking.
        const int to_q = __builtin_amdgcn_readfirstlane(
            (MODE == 0 && (ncol0 + jj * 16 < 256)) ? 1 : 0);
#pragma unroll
        for (int r = 0; r < 8; ++r) {
            const int R   = gm0 + wm * 16 + hi * 8 + r;   // D row = hi*8+r
            const int col = ncol0 + jj * 16 + n;
            const float val = (jj ? acc1[r] : acc0[r]) + bias[col];
            if (MODE == 0) {
                if (to_q) q_bf[(size_t)R * 256 + col]       = f2bf(val);
                else      k_bf[(size_t)R * 256 + col - 256] = f2bf(val);
            } else {
                out_f32[(size_t)R * 256 + col] = val;
            }
        }
    }
}

// ---------------------------------------------------------------------------
// Flash attention for one cross-shaped window instance.
// VERT=0: horizontal stripes (channels [0,128)), token t = r*64+w.
// VERT=1: vertical stripes   (channels [128,256)), token t = y*8+s,
//         with the (intentionally) scrambled inverse mapping of the reference.
// One wave = 16 query rows; per-wave LDS slices, double-buffered async K/V
// staging overlapped with WMMA + softmax of the current block.
// ---------------------------------------------------------------------------
template<int VERT>
__device__ __forceinline__ int pix_of(int b, int win, int t) {
    if (VERT == 0) return ((b * 64 + win * 8 + (t >> 6)) << 6) + (t & 63);
    return ((b * 64 + (t >> 3)) << 6) + win * 8 + (t & 7);
}

template<int VERT>
__global__ __launch_bounds__(256) void attn_wmma_kernel(
    const unsigned short* __restrict__ q,
    const unsigned short* __restrict__ k,   // also V
    float* __restrict__ ctx)
{
    // [buf][wave][32 keys x 40 (padded) bf16]
    __shared__ __align__(16) unsigned short kbuf[2][8][32 * 40];
    __shared__ __align__(16) unsigned short pbuf[8][16 * 32];

    const int wid  = threadIdx.x >> 5;
    const int lane = threadIdx.x & 31;
    const int n    = lane & 15, hi = lane >> 4;

    const int inst = blockIdx.x >> 2;
    const int qblk = blockIdx.x & 3;
    const int b    = inst >> 5;
    const int rem  = inst & 31;
    const int win  = rem >> 2;
    const int head = rem & 3;
    const int chb  = (VERT ? 128 : 0) + head * 32;
    const int qt0  = qblk * 128 + wid * 16;

    unsigned short* pb = pbuf[wid];
    // Per-lane LDS byte addresses of this wave's two staging rows.
    const unsigned lds_row[2] = {
        (unsigned)(uintptr_t)(const void*)(&kbuf[0][wid][0]) + (unsigned)lane * 80u,
        (unsigned)(uintptr_t)(const void*)(&kbuf[1][wid][0]) + (unsigned)lane * 80u
    };

    // Q fragment (16 queries x d=32), A layout
    BFrag qf;
    {
        const unsigned short* ap =
            q + (size_t)pix_of<VERT>(b, win, qt0 + n) * 256 + chb;
        qf.h[0] = *(const v8u16*)(ap + hi * 16);
        qf.h[1] = *(const v8u16*)(ap + hi * 16 + 8);
    }

    float mrow[8], lrow[8];
#pragma unroll
    for (int r = 0; r < 8; ++r) { mrow[r] = -1e30f; lrow[r] = 0.0f; }
    v8f ctx0 = {}, ctx1 = {};
    const float scale = 0.1767766952966369f;   // 1/sqrt(32)

    // Prefetch key block 0 into buffer 0 (lane -> key kb0+lane, 64B row).
    stage_async64(lds_row[0],
                  k + (size_t)pix_of<VERT>(b, win, lane) * 256 + chb);

#pragma unroll 1
    for (int it = 0; it < 16; ++it) {
        const int kb0 = it * 32;
        unsigned short* kb = &kbuf[it & 1][wid][0];

        if (it < 15) {
            // Issue next block into the alternate buffer, then wait until
            // only those 4 async loads remain in flight (in-order completion
            // => current buffer is ready).
            stage_async64(lds_row[(it & 1) ^ 1],
                          k + (size_t)pix_of<VERT>(b, win, kb0 + 32 + lane) * 256 + chb);
            wait_async_keep4();
        } else {
            wait_async_all();
        }

        // K^T fragments: B layout, lane n = key column, elems = d = hi*16+e.
        BFrag kt0, kt1;
        kt0.h[0] = *(const v8u16*)(kb + n * 40 + hi * 16);
        kt0.h[1] = *(const v8u16*)(kb + n * 40 + hi * 16 + 8);
        kt1.h[0] = *(const v8u16*)(kb + (16 + n) * 40 + hi * 16);
        kt1.h[1] = *(const v8u16*)(kb + (16 + n) * 40 + hi * 16 + 8);

        v8f z = {};
        v8f s0 = __builtin_amdgcn_wmma_f32_16x16x32_bf16(
                     false, qf.b, false, kt0.b, (short)0, z, false, false);
        v8f s1 = __builtin_amdgcn_wmma_f32_16x16x32_bf16(
                     false, qf.b, false, kt1.b, (short)0, z, false, false);

        float tm[8], p0[8], p1[8], alpha[8], rs[8];
#pragma unroll
        for (int r = 0; r < 8; ++r) {
            s0[r] *= scale; s1[r] *= scale;
            tm[r] = fmaxf(s0[r], s1[r]);
        }
        // Row reductions stay inside the 16-lane half-group that holds rows
        // hi*8 .. hi*8+7 (xor masks 1,2,4,8 never cross the half boundary).
#pragma unroll
        for (int m = 1; m < 16; m <<= 1)
#pragma unroll
            for (int r = 0; r < 8; ++r)
                tm[r] = fmaxf(tm[r], __shfl_xor(tm[r], m, 32));
#pragma unroll
        for (int r = 0; r < 8; ++r) {
            const float mn = fmaxf(mrow[r], tm[r]);
            alpha[r] = __expf(mrow[r] - mn);
            mrow[r]  = mn;
            p0[r] = __expf(s0[r] - mn);
            p1[r] = __expf(s1[r] - mn);
            rs[r] = p0[r] + p1[r];
        }
#pragma unroll
        for (int m = 1; m < 16; m <<= 1)
#pragma unroll
            for (int r = 0; r < 8; ++r)
                rs[r] += __shfl_xor(rs[r], m, 32);
#pragma unroll
        for (int r = 0; r < 8; ++r) {
            lrow[r] = lrow[r] * alpha[r] + rs[r];
            ctx0[r] *= alpha[r];
            ctx1[r] *= alpha[r];
            // P tile (D layout) -> LDS in (query, key) order for A-frag reload
            pb[(hi * 8 + r) * 32 + n]      = f2bf(p0[r]);
            pb[(hi * 8 + r) * 32 + 16 + n] = f2bf(p1[r]);
        }
        BFrag pf;
        pf.h[0] = *(const v8u16*)(pb + n * 32 + hi * 16);
        pf.h[1] = *(const v8u16*)(pb + n * 32 + hi * 16 + 8);

        // V fragments: B layout, lane n = output d column, elems = key hi*16+e
        BFrag vf0, vf1;
#pragma unroll
        for (int e = 0; e < 16; ++e) {
            vf0.u[e] = kb[(hi * 16 + e) * 40 + n];
            vf1.u[e] = kb[(hi * 16 + e) * 40 + 16 + n];
        }
        ctx0 = __builtin_amdgcn_wmma_f32_16x16x32_bf16(
                   false, pf.b, false, vf0.b, (short)0, ctx0, false, false);
        ctx1 = __builtin_amdgcn_wmma_f32_16x16x32_bf16(
                   false, pf.b, false, vf1.b, (short)0, ctx1, false, false);
    }

#pragma unroll
    for (int r = 0; r < 8; ++r) {
        const float inv = 1.0f / lrow[r];
        const int   t   = qt0 + hi * 8 + r;
        const float c0  = ctx0[r] * inv;
        const float c1  = ctx1[r] * inv;
        if (VERT == 0) {
            const size_t p = (size_t)pix_of<0>(b, win, t) * 256 + head * 32;
            ctx[p + n]      = c0;
            ctx[p + 16 + n] = c1;
        } else {
            // Scrambled inverse of the reference's TF-order reshape chain.
            const int t64 = t & 63;
            const int H_  = win * 8 + (t >> 6);
            const int W_  = head * 16 + (t64 >> 2);
            const size_t p =
                ((size_t)(b * 64 + H_) * 64 + W_) * 256 + 128 + (t64 & 3) * 32;
            ctx[p + n]      = c0;
            ctx[p + 16 + n] = c1;
        }
    }
}

// ---------------------------------------------------------------------------
// Depthwise 3x3 'SAME' conv on hard_swish(v) + residual add + hard_swish,
// emitting bf16 activations for the final WMMA projection.
// One block per pixel, one thread per channel.
// ---------------------------------------------------------------------------
__global__ __launch_bounds__(256) void dwconv_act_kernel(
    const unsigned short* __restrict__ v_bf, const float* __restrict__ dwk,
    const float* __restrict__ dwb, const float* __restrict__ ctx_in,
    unsigned short* __restrict__ ctx_bf)
{
    const int p = blockIdx.x;
    const int c = threadIdx.x;
    const int x = p & 63;
    const int y = (p >> 6) & 63;
    const int b = p >> 12;
    float acc = dwb[c];
#pragma unroll
    for (int kh = 0; kh < 3; ++kh) {
        const int yy = y + kh - 1;
        if (yy < 0 || yy > 63) continue;
#pragma unroll
        for (int kw = 0; kw < 3; ++kw) {
            const int xx = x + kw - 1;
            if (xx < 0 || xx > 63) continue;
            const float v =
                bf2f(v_bf[(((size_t)(b * 64 + yy)) * 64 + xx) * 256 + c]);
            acc += hswish(v) * dwk[(kh * 3 + kw) * 256 + c];  // (3,3,C,1)
        }
    }
    const float o = hswish(ctx_in[(size_t)p * 256 + c] + acc);
    ctx_bf[(size_t)p * 256 + c] = f2bf(o);
}

// ---------------------------------------------------------------------------
// Host-side orchestration.
// Workspace layout (bytes):
//   [0,16M)      q_bf16   32768x256
//   [16M,32M)    k_bf16   32768x256 (also V)
//   [32M,64M)    ctx_f32  32768x256
//   [64M,80M)    ctx_bf16 32768x256
//   [80M,+256K)  w_qkv swizzled bf16
//   [.. ,+128K)  w_out swizzled bf16          total ~80.4 MB
// ---------------------------------------------------------------------------
extern "C" void kernel_launch(void* const* d_in, const int* in_sizes, int n_in,
                              void* d_out, int out_size, void* d_ws, size_t ws_size,
                              hipStream_t stream) {
    (void)in_sizes; (void)n_in; (void)out_size; (void)ws_size;

    const float* x      = (const float*)d_in[0];
    const float* w_qkv  = (const float*)d_in[1];
    const float* b_qkv  = (const float*)d_in[2];
    const float* dwk    = (const float*)d_in[3];
    const float* dwb    = (const float*)d_in[4];
    const float* w_out  = (const float*)d_in[5];
    const float* b_out  = (const float*)d_in[6];

    char* ws = (char*)d_ws;
    unsigned short* q_bf    = (unsigned short*)(ws + 0);
    unsigned short* k_bf    = (unsigned short*)(ws + (size_t)16777216);
    float*          ctx_f   = (float*)         (ws + (size_t)33554432);
    unsigned short* ctx_bf  = (unsigned short*)(ws + (size_t)67108864);
    unsigned short* wqkv_sw = (unsigned short*)(ws + (size_t)83886080);
    unsigned short* wout_sw = (unsigned short*)(ws + (size_t)84148224);

    // 1) weight conversion into WMMA B-fragment order
    swizzle_w_kernel<<<(256 * 512 + 255) / 256, 256, 0, stream>>>(
        w_qkv, wqkv_sw, 256, 512);
    swizzle_w_kernel<<<(256 * 256 + 255) / 256, 256, 0, stream>>>(
        w_out, wout_sw, 256, 256);

    // 2) qkv projection: M=32768, K=256, N=512 -> q_bf16 / k_bf16
    gemm_wmma_kernel<0, 0><<<dim3(8, 512), 256, 0, stream>>>(
        (const void*)x, wqkv_sw, b_qkv, 256, 512, q_bf, k_bf, nullptr);

    // 3) horizontal + vertical stripe attention (256 instances x 4 blocks)
    attn_wmma_kernel<0><<<1024, 256, 0, stream>>>(q_bf, k_bf, ctx_f);
    attn_wmma_kernel<1><<<1024, 256, 0, stream>>>(q_bf, k_bf, ctx_f);

    // 4) depthwise conv bias branch + activation -> bf16 activations
    dwconv_act_kernel<<<32768, 256, 0, stream>>>(k_bf, dwk, dwb, ctx_f, ctx_bf);

    // 5) output projection: M=32768, K=256, N=256 -> f32 out
    gemm_wmma_kernel<1, 1><<<dim3(4, 512), 256, 0, stream>>>(
        (const void*)ctx_bf, wout_sw, b_out, 256, 256, nullptr, nullptr,
        (float*)d_out);
}
